// LeechAttention_2336462209660
// MI455X (gfx1250) — matrix operand order repaired and checked by reading, not compiled, not run on hardware
//
#include <hip/hip_runtime.h>

#define DI __device__ __forceinline__

typedef unsigned short u16;
typedef __attribute__((ext_vector_type(16))) __bf16 bf16x16;
typedef __attribute__((ext_vector_type(8)))  float   f32x8;

union FragU { uint4 u[2]; bf16x16 v; };

namespace {
constexpr int Bb   = 4;
constexpr int Tt   = 2048;
constexpr int Dd   = 768;
constexpr int Hh   = 8;
constexpr int HDd  = 96;
constexpr int Mrows = Bb * Tt;     // 8192
constexpr int Eqkv  = 3 * Dd;      // 2304
constexpr float SCALE  = 0.10206207261596576f;  // 96^-0.5
constexpr float NEGINF = -1e30f;
}

DI u16 f2bf(float f) {
  unsigned u = __builtin_bit_cast(unsigned, f);
  u += 0x7FFFu + ((u >> 16) & 1u);       // round-to-nearest-even
  return (u16)(u >> 16);
}

DI f32x8 fzero() {
  f32x8 z = {0.f, 0.f, 0.f, 0.f, 0.f, 0.f, 0.f, 0.f};
  return z;
}

DI f32x8 wmma_bf16(const FragU& a, const FragU& b, f32x8 c) {
  return __builtin_amdgcn_wmma_f32_16x16x32_bf16(false, a.v, false, b.v,
                                                 (short)0, c, false, false);
}

// ---------------------------------------------------------------------------
// Kernel 1a: f32 -> bf16 cast
// ---------------------------------------------------------------------------
__global__ void cast_f32_to_bf16(const float* __restrict__ src,
                                 u16* __restrict__ dst, int n) {
  int i = blockIdx.x * blockDim.x + threadIdx.x;
  if (i < n) dst[i] = f2bf(src[i]);
}

// ---------------------------------------------------------------------------
// Kernel 1b: fold 24x24 Leech rotation into q/k rows of w_qkv, cast to bf16.
// q' = q @ Kblk  =>  W'[e(j)] = sum_i kern[i][j] * W[e(i)] within each block.
// ---------------------------------------------------------------------------
__global__ void rot_wqkv_bf16(const float* __restrict__ w,
                              const float* __restrict__ kern,
                              u16* __restrict__ dst) {
  int idx = blockIdx.x * blockDim.x + threadIdx.x;
  if (idx >= Eqkv * Dd) return;
  int e = idx / Dd;
  int d = idx - e * Dd;
  int s = e / Dd;                 // 0=q, 1=k, 2=v
  float val;
  if (s == 2) {
    val = w[e * Dd + d];
  } else {
    int r  = e - s * Dd;          // 0..767
    int hd = r % HDd;
    int j  = hd % 24;
    int ebase = e - j;            // first row of this 24-block
    float acc = 0.f;
    for (int i = 0; i < 24; ++i)
      acc = fmaf(kern[i * 24 + j], w[(ebase + i) * Dd + d], acc);
    val = acc;
  }
  dst[e * Dd + d] = f2bf(val);
}

// ---------------------------------------------------------------------------
// Kernel 2: QKV GEMM  qkv[m,e] = sum_d xb[m,d]*wqb[e,d]   (bf16 WMMA, f32 acc)
// One wave computes a 16x64 output strip. Scatters q,k as [BH,T,96] and v
// transposed as [BH,96,T] (so attention's PV B-fragments load contiguously).
// ---------------------------------------------------------------------------
__global__ __launch_bounds__(256) void qkv_gemm(
    const u16* __restrict__ xb, const u16* __restrict__ wqb,
    u16* __restrict__ qb, u16* __restrict__ kbuf, u16* __restrict__ vtb) {
  const int lane = threadIdx.x & 31;
  const int wv   = (blockIdx.x * blockDim.x + threadIdx.x) >> 5;
  const int NSTRIP = Eqkv / 64;                 // 36
  const int mt = wv / NSTRIP;
  const int st = wv - mt * NSTRIP;
  if (mt >= Mrows / 16) return;
  const int ln = lane & 15, half = lane >> 4;

  const u16* arow = xb + (mt * 16 + ln) * Dd;
  f32x8 acc[4] = {fzero(), fzero(), fzero(), fzero()};

  for (int kb = 0; kb < Dd; kb += 32) {
    FragU a;   // A 16x32: lane=row, K run of 16 per half
    a.u[0] = *(const uint4*)(arow + kb + half * 16);
    a.u[1] = *(const uint4*)(arow + kb + half * 16 + 8);
#pragma unroll
    for (int nt = 0; nt < 4; ++nt) {
      const u16* brow = wqb + (st * 64 + nt * 16 + ln) * Dd + kb + half * 16;
      FragU b;  // B 32x16: lane=column, K run of 16 per half
      b.u[0] = ((const uint4*)brow)[0];
      b.u[1] = ((const uint4*)brow)[1];
      acc[nt] = wmma_bf16(a, b, acc[nt]);
    }
  }

#pragma unroll
  for (int nt = 0; nt < 4; ++nt) {
    int e  = st * 64 + nt * 16 + ln;
    int s  = e / Dd;
    int r2 = e - s * Dd;
    int h  = r2 / HDd;
    int hd = r2 - h * HDd;
#pragma unroll
    for (int r = 0; r < 8; ++r) {
      int m = mt * 16 + r + half * 8;            // C layout: row = r + 8*half
      int b = m >> 11, t = m & (Tt - 1);
      int bh = b * Hh + h;
      u16 val = f2bf(acc[nt][r]);
      if (s == 0)      qb[(bh * Tt + t) * HDd + hd] = val;
      else if (s == 1) kbuf[(bh * Tt + t) * HDd + hd] = val;
      else             vtb[(bh * HDd + hd) * Tt + t] = val;
    }
  }
}

// ---------------------------------------------------------------------------
// Kernel 3: causal flash attention, one wave per 16-row q tile, 32 keys/step.
// ---------------------------------------------------------------------------
__global__ __launch_bounds__(256) void attn_fwd(
    const u16* __restrict__ qg, const u16* __restrict__ kg,
    const u16* __restrict__ vtg, u16* __restrict__ ag) {
  __shared__ u16 lds[8 * 512];                  // 16x32 bf16 P tile per wave
  const int lane  = threadIdx.x & 31;
  const int wloc  = threadIdx.x >> 5;
  const int wbase = wloc * 512;
  const int gw    = blockIdx.x * 8 + wloc;
  const int bh    = gw >> 7;                    // 0..31
  const int qt    = gw & 127;
  const int qbase = qt * 16;
  const int ln = lane & 15, half = lane >> 4;

  // Q fragments (A layout): rows qbase+ln, K = 96 as 3x32
  FragU qa[3];
  const u16* qrow = qg + (bh * Tt + qbase + ln) * HDd;
#pragma unroll
  for (int t3 = 0; t3 < 3; ++t3) {
    qa[t3].u[0] = *(const uint4*)(qrow + t3 * 32 + half * 16);
    qa[t3].u[1] = *(const uint4*)(qrow + t3 * 32 + half * 16 + 8);
  }

  float mr[8], lr[8], alr[8];
  f32x8 o[6];
#pragma unroll
  for (int r = 0; r < 8; ++r) { mr[r] = NEGINF; lr[r] = 0.f; }
#pragma unroll
  for (int c = 0; c < 6; ++c) o[c] = fzero();

  for (int j0 = 0; j0 <= qbase + 15; j0 += 32) {
    // S = Q K^T for keys [j0, j0+32)
    f32x8 s0 = fzero(), s1 = fzero();
#pragma unroll
    for (int t3 = 0; t3 < 3; ++t3) {
      const u16* kr0 = kg + (bh * Tt + j0 + ln) * HDd + t3 * 32 + half * 16;
      FragU b0; b0.u[0] = ((const uint4*)kr0)[0]; b0.u[1] = ((const uint4*)kr0)[1];
      s0 = wmma_bf16(qa[t3], b0, s0);
      const u16* kr1 = kr0 + 16 * HDd;
      FragU b1; b1.u[0] = ((const uint4*)kr1)[0]; b1.u[1] = ((const uint4*)kr1)[1];
      s1 = wmma_bf16(qa[t3], b1, s1);
    }

    // online softmax in C layout (VGPR r -> row r+8*half, lane -> column)
#pragma unroll
    for (int r = 0; r < 8; ++r) {
      int row = qbase + r + half * 8;
      int c0 = j0 + ln, c1 = c0 + 16;
      float v0 = (c0 <= row) ? s0[r] * SCALE : NEGINF;
      float v1 = (c1 <= row) ? s1[r] * SCALE : NEGINF;
      float cm = fmaxf(v0, v1);
      cm = fmaxf(cm, __shfl_xor(cm, 1, 32));
      cm = fmaxf(cm, __shfl_xor(cm, 2, 32));
      cm = fmaxf(cm, __shfl_xor(cm, 4, 32));
      cm = fmaxf(cm, __shfl_xor(cm, 8, 32));
      float nm = fmaxf(mr[r], cm);
      float al = __expf(mr[r] - nm);
      v0 = __expf(v0 - nm);
      v1 = __expf(v1 - nm);
      float rs = v0 + v1;
      rs += __shfl_xor(rs, 1, 32);
      rs += __shfl_xor(rs, 2, 32);
      rs += __shfl_xor(rs, 4, 32);
      rs += __shfl_xor(rs, 8, 32);
      lr[r] = lr[r] * al + rs;
      mr[r] = nm;
      alr[r] = al;
      lds[wbase + (r + half * 8) * 32 + ln]      = f2bf(v0);
      lds[wbase + (r + half * 8) * 32 + ln + 16] = f2bf(v1);
    }
#pragma unroll
    for (int c = 0; c < 6; ++c)
#pragma unroll
      for (int r = 0; r < 8; ++r) o[c][r] *= alr[r];

    // wave-private LDS relayout C->A; LDS ops are in-order per wave,
    // enforce completion before re-reading as A fragments.
    asm volatile("s_wait_dscnt 0" ::: "memory");

    FragU pa;  // P as A 16x32 (lane = query row, K = key index)
    pa.u[0] = *(const uint4*)&lds[wbase + ln * 32 + half * 16];
    pa.u[1] = *(const uint4*)&lds[wbase + ln * 32 + half * 16 + 8];

    // O += P @ V  (V stored transposed [BH,96,T]: contiguous B loads)
#pragma unroll
    for (int c = 0; c < 6; ++c) {
      const u16* vr = vtg + (bh * HDd + c * 16 + ln) * Tt + j0 + half * 16;
      FragU vb; vb.u[0] = ((const uint4*)vr)[0]; vb.u[1] = ((const uint4*)vr)[1];
      o[c] = wmma_bf16(pa, vb, o[c]);
    }
  }

  const int b = bh >> 3, h = bh & 7;
#pragma unroll
  for (int c = 0; c < 6; ++c)
#pragma unroll
    for (int r = 0; r < 8; ++r) {
      int row = qbase + r + half * 8;
      float val = o[c][r] / lr[r];
      ag[(b * Tt + row) * Dd + h * HDd + c * 16 + ln] = f2bf(val);
    }
}

// ---------------------------------------------------------------------------
// Kernel 4: output GEMM  out[m,n] = sum_d ag[m,d]*wob[n,d], f32 result
// ---------------------------------------------------------------------------
__global__ __launch_bounds__(256) void out_gemm(
    const u16* __restrict__ ab, const u16* __restrict__ wob,
    float* __restrict__ out) {
  const int lane = threadIdx.x & 31;
  const int wv   = (blockIdx.x * blockDim.x + threadIdx.x) >> 5;
  const int NSTRIP = Dd / 64;                   // 12
  const int mt = wv / NSTRIP;
  const int st = wv - mt * NSTRIP;
  if (mt >= Mrows / 16) return;
  const int ln = lane & 15, half = lane >> 4;

  const u16* arow = ab + (mt * 16 + ln) * Dd;
  f32x8 acc[4] = {fzero(), fzero(), fzero(), fzero()};

  for (int kb = 0; kb < Dd; kb += 32) {
    FragU a;
    a.u[0] = *(const uint4*)(arow + kb + half * 16);
    a.u[1] = *(const uint4*)(arow + kb + half * 16 + 8);
#pragma unroll
    for (int nt = 0; nt < 4; ++nt) {
      const u16* brow = wob + (st * 64 + nt * 16 + ln) * Dd + kb + half * 16;
      FragU b;
      b.u[0] = ((const uint4*)brow)[0];
      b.u[1] = ((const uint4*)brow)[1];
      acc[nt] = wmma_bf16(a, b, acc[nt]);
    }
  }

#pragma unroll
  for (int nt = 0; nt < 4; ++nt) {
    int n = st * 64 + nt * 16 + ln;
#pragma unroll
    for (int r = 0; r < 8; ++r) {
      int m = mt * 16 + r + half * 8;
      out[m * Dd + n] = acc[nt][r];
    }
  }
}

// ---------------------------------------------------------------------------
extern "C" void kernel_launch(void* const* d_in, const int* in_sizes, int n_in,
                              void* d_out, int out_size, void* d_ws,
                              size_t ws_size, hipStream_t stream) {
  (void)in_sizes; (void)n_in; (void)out_size; (void)ws_size;
  const float* x    = (const float*)d_in[0];   // [4,2048,768]
  const float* wqkv = (const float*)d_in[1];   // [2304,768]
  const float* wout = (const float*)d_in[2];   // [768,768]
  const float* kern = (const float*)d_in[3];   // [24,24]
  float* out = (float*)d_out;

  char* ws = (char*)d_ws;
  u16* xb   = (u16*)(ws);                         // 8192*768   bf16
  u16* wqb  = (u16*)(ws + 12582912);              // 2304*768   bf16 (rotated)
  u16* wob  = (u16*)(ws + 16121856);              // 768*768    bf16
  u16* qb   = (u16*)(ws + 17301504);              // [32,2048,96]
  u16* kbuf = (u16*)(ws + 29884416);              // [32,2048,96]
  u16* vtb  = (u16*)(ws + 42467328);              // [32,96,2048] (transposed)
  u16* abuf = (u16*)(ws + 55050240);              // 8192*768 attn out bf16
                                                  // total 67,633,152 B

  // Stage 1: casts + weight-folded Leech rotation
  {
    int n = Mrows * Dd;
    cast_f32_to_bf16<<<(n + 255) / 256, 256, 0, stream>>>(x, xb, n);
  }
  {
    int n = Dd * Dd;
    cast_f32_to_bf16<<<(n + 255) / 256, 256, 0, stream>>>(wout, wob, n);
  }
  {
    int n = Eqkv * Dd;
    rot_wqkv_bf16<<<(n + 255) / 256, 256, 0, stream>>>(wqkv, kern, wqb);
  }

  // Stage 2: QKV projection -> q, k, v^T (bf16)
  {
    int waves = (Mrows / 16) * (Eqkv / 64);       // 512*36 = 18432
    qkv_gemm<<<waves / 8, 256, 0, stream>>>(xb, wqb, qb, kbuf, vtb);
  }

  // Stage 3: causal flash attention
  {
    int waves = Bb * Hh * (Tt / 16);              // 4096
    attn_fwd<<<waves / 8, 256, 0, stream>>>(qb, kbuf, vtb, abuf);
  }

  // Stage 4: output projection (f32 result)
  {
    int waves = (Mrows / 16) * (Dd / 64);         // 512*12 = 6144
    out_gemm<<<waves / 8, 256, 0, stream>>>(abuf, wob, out);
  }
}
